// WeatherModel_5729486373382
// MI455X (gfx1250) — compile-verified, matrix operand order checked by reading, not run on hardware
//
#include <hip/hip_runtime.h>
#include <hip/hip_bf16.h>

// ---------------- problem constants ----------------
#define WZ 2
#define WH 12
#define WWIN 6
#define ZD 8
#define HD 180
#define W0D 91
#define WPD 96
#define CD 192
#define HEADS 6
#define HS 32
#define TT 144           // tokens per window
#define NWZ 4
#define NWH 15
#define NWW 16
#define NW  960          // windows
#define M_ATT (NW * TT)  // 138240
#define TOK (ZD * HD * W0D) // 131040
#define SCALEF 0.17677669529663687f

typedef _Float16 v16h __attribute__((ext_vector_type(16)));
typedef _Float16 v8h  __attribute__((ext_vector_type(8)));
typedef float    v8f  __attribute__((ext_vector_type(8)));

// ---------------- WMMA fragment loader ----------------
// A/B 16-bit fragment layout (ISA 7.12.2): lane%16 = row (A) / col (B),
// half = lane/16 : elements e<8 -> k = half*8+e ; e>=8 -> k = 16+half*8+(e-8)
static __device__ __forceinline__ v16h load_frag(const _Float16* base, int row, int half, int stride) {
  const _Float16* p = base + (size_t)row * stride + half * 8;
  v8h x0 = *(const v8h*)p;
  v8h x1 = *(const v8h*)(p + 16);
  v16h r;
#pragma unroll
  for (int i = 0; i < 8; ++i) { r[i] = x0[i]; r[i + 8] = x1[i]; }
  return r;
}

static __device__ __forceinline__ v8f wmma_f16(v16h a, v16h b, v8f c) {
  return __builtin_amdgcn_wmma_f32_16x16x32_f16(false, a, false, b, (short)0, c, false, false);
}

// ---------------- f32 -> f16 convert ----------------
__global__ void cvt_kernel(const float* __restrict__ in, _Float16* __restrict__ out, int n) {
  int i = blockIdx.x * blockDim.x + threadIdx.x;
  if (i < n) out[i] = (_Float16)in[i];
}

// ---------------- gather: pad + roll + window partition (8 ch / thread) ----------------
__global__ void gather_kernel(const float* __restrict__ x, _Float16* __restrict__ xw,
                              const int* __restrict__ rollp) {
  long long e = (long long)blockIdx.x * blockDim.x + threadIdx.x;
  if (e >= (long long)NW * TT * (CD / 8)) return;
  int cb = (int)(e % (CD / 8));
  int t  = (int)((e / (CD / 8)) % TT);
  int w  = (int)(e / ((long long)(CD / 8) * TT));
  int roll = rollp[0];
  int z_in = t / (WH * WWIN), h_in = (t % (WH * WWIN)) / WWIN, w_in = t % WWIN;
  int wz = w / (NWH * NWW), rem = w % (NWH * NWW), wh = rem / NWW, ww = rem % NWW;
  int zp = wz * WZ + z_in, hp = wh * WH + h_in, wp = ww * WWIN + w_in;
  int zs, hs, ws;
  if (roll) { zs = (zp + ZD - 1) % ZD; hs = (hp + HD - 6) % HD; ws = (wp + WPD - 3) % WPD; }
  else      { zs = zp; hs = hp; ws = wp; }
  int w0 = ws - 3;
  v8h out = {};
  if (w0 >= 0 && w0 < W0D) {
    const float* src = x + (((size_t)zs * HD + hs) * W0D + w0) * CD + cb * 8;
#pragma unroll
    for (int i = 0; i < 8; ++i) out[i] = (_Float16)src[i];
  }
  *(v8h*)(xw + ((size_t)w * TT + t) * CD + cb * 8) = out;
}

// ---------------- generic WMMA GEMM: Out = A(MxK) * W(NxK)^T + bias ----------------
// EPI: 0 = f16 out, 1 = f32 out, 2 = f16 out with exact GELU
// block 256 = 8 waves (4M x 2N); block tile 128M x 64N; K-step 32, reg double-buffered
template <int EPI>
__global__ __launch_bounds__(256) void gemm_wmma(const _Float16* __restrict__ A,
                                                 const _Float16* __restrict__ W,
                                                 const float* __restrict__ bias,
                                                 void* __restrict__ Out,
                                                 int M, int N, int K) {
  __shared__ __align__(16) _Float16 As[128 * 32];
  __shared__ __align__(16) _Float16 Bs[64 * 32];
  int n0 = blockIdx.x * 64;
  int m0 = blockIdx.y * 128;
  int tid = threadIdx.x;
  int lane = tid & 31, wave = tid >> 5;
  int wm = wave & 3, wn = wave >> 2;     // 4 (M) x 2 (N) wave grid
  int r16 = lane & 15, hf = lane >> 4;
  v8f acc[2][2] = {{{}, {}}, {{}, {}}};

  int rowA = tid >> 1, chA = tid & 1;    // 128 rows x 2 chunks of 16
  int rowB = tid >> 2, chB = tid & 3;    // 64 rows x 4 chunks of 8
  bool okA = (m0 + rowA) < M;
  const _Float16* gA = A + (size_t)(m0 + rowA) * K + chA * 16;
  const _Float16* gB = W + (size_t)(n0 + rowB) * K + chB * 8;

  v8h ra0 = {}, ra1 = {}, rb0 = {};
  if (okA) { ra0 = *(const v8h*)gA; ra1 = *(const v8h*)(gA + 8); }
  rb0 = *(const v8h*)gB;

  int KT = K / 32;
  for (int kt = 0; kt < KT; ++kt) {
    *(v8h*)(As + rowA * 32 + chA * 16)     = ra0;
    *(v8h*)(As + rowA * 32 + chA * 16 + 8) = ra1;
    *(v8h*)(Bs + rowB * 32 + chB * 8)      = rb0;
    __syncthreads();
    if (kt + 1 < KT) {                    // prefetch next K-step during WMMAs
      const _Float16* pa = gA + (kt + 1) * 32;
      if (okA) { ra0 = *(const v8h*)pa; ra1 = *(const v8h*)(pa + 8); }
      rb0 = *(const v8h*)(gB + (kt + 1) * 32);
    }
    v16h a0 = load_frag(As + (wm * 32 + 0) * 32, r16, hf, 32);
    v16h a1 = load_frag(As + (wm * 32 + 16) * 32, r16, hf, 32);
    v16h b0 = load_frag(Bs + (wn * 32 + 0) * 32, r16, hf, 32);
    v16h b1 = load_frag(Bs + (wn * 32 + 16) * 32, r16, hf, 32);
    acc[0][0] = wmma_f16(a0, b0, acc[0][0]);
    acc[0][1] = wmma_f16(a0, b1, acc[0][1]);
    acc[1][0] = wmma_f16(a1, b0, acc[1][0]);
    acc[1][1] = wmma_f16(a1, b1, acc[1][1]);
    __syncthreads();
  }

#pragma unroll
  for (int im = 0; im < 2; ++im)
#pragma unroll
    for (int in = 0; in < 2; ++in) {
      int col = n0 + wn * 32 + in * 16 + r16;
      float bv = bias[col];
#pragma unroll
      for (int e = 0; e < 8; ++e) {
        int rowg = m0 + wm * 32 + im * 16 + e + 8 * hf;
        if (rowg < M) {
          float v = acc[im][in][e] + bv;
          if (EPI == 2) v = 0.5f * v * (1.0f + erff(v * 0.70710678118654752f));
          if (EPI == 1) ((float*)Out)[(size_t)rowg * N + col] = v;
          else          ((_Float16*)Out)[(size_t)rowg * N + col] = (_Float16)v;
        }
      }
    }
}

// ---------------- attention ----------------
static __device__ __forceinline__ int region_of(int zp, int hp, int wp) {
  int zc = (zp < ZD - WZ) ? 0 : ((zp < ZD - 1) ? 1 : 2);
  int hc = (hp < HD - WH) ? 0 : ((hp < HD - 6) ? 1 : 2);
  int wc = (wp < WPD - WWIN) ? 0 : ((wp < WPD - 3) ? 1 : 2);
  return zc * 9 + hc * 3 + wc;
}

// grid = NW * HEADS * 3 (48-row chunks), block = 128 (4 waves)
__global__ __launch_bounds__(128) void attn_kernel(const _Float16* __restrict__ qkv,
                                                   const float* __restrict__ btab,
                                                   _Float16* __restrict__ attn_out,
                                                   const int* __restrict__ rollp) {
  const int SSTR = 152;
  __shared__ __align__(16) float S[48 * SSTR];        // scores (f32)
  __shared__ __align__(16) _Float16 Pf[48 * 160];     // softmax probs (f16, K padded)
  __shared__ __align__(16) _Float16 Vt[HS * 160];     // V^T (f16, K padded)
  __shared__ unsigned char tz[TT], ty[TT], tx[TT], treg[TT];

  int bid = blockIdx.x;
  int chunk = bid % 3;
  int whid = bid / 3;
  int h = whid % HEADS;
  int w = whid / HEADS;
  int wz = w / (NWH * NWW), rem = w % (NWH * NWW), wh = rem / NWW, ww = rem % NWW;
  int roll = rollp[0];
  int tid = threadIdx.x, lane = tid & 31, wave = tid >> 5;
  int r16 = lane & 15, hf = lane >> 4;
  int m0 = chunk * 48;

  const _Float16* qbase = qkv + (size_t)w * TT * 576 + h * HS;
  const _Float16* kbase = qbase + CD;
  const _Float16* vbase = qbase + 2 * CD;

  // ---- per-token coordinate / region tables ----
  if (tid < TT) {
    int t = tid;
    int z1 = t / (WH * WWIN), h1 = (t % (WH * WWIN)) / WWIN, w1 = t % WWIN;
    tz[t] = (unsigned char)z1;
    ty[t] = (unsigned char)h1;
    tx[t] = (unsigned char)w1;
    treg[t] = (unsigned char)region_of(wz * WZ + z1, wh * WH + h1, ww * WWIN + w1);
  }
  // ---- stage V^T into LDS (zero-padded K to 160) ----
  for (int i = tid; i < TT * HS; i += 128) {
    int k = i >> 5, d = i & 31;
    Vt[d * 160 + k] = vbase[(size_t)k * 576 + d];
  }
  for (int i = tid; i < HS * 16; i += 128) {
    int d = i >> 4, j = i & 15;
    Vt[d * 160 + TT + j] = (_Float16)0.f;
  }
  __syncthreads();

  // ---- init S with pos_bias + shift_mask ----
  int woff = wz * NWW + ww;
  for (int i = tid; i < 48 * TT; i += 128) {
    int m = i / TT, n = i - m * TT;
    int t = m0 + m;
    int idx = (((int)tz[t] + (int)tz[n] * WZ) * 23 + ((int)ty[t] - (int)ty[n] + WH - 1)) * 36 +
              ((int)tx[t] + (int)tx[n] * WWIN);
    float pb = btab[((size_t)idx * (NWZ * NWW) + woff) * HEADS + h];
    float mk = (roll && treg[t] != treg[n]) ? -1000.f : 0.f;
    S[m * SSTR + n] = pb + mk;
  }
  __syncthreads();

  // ---- phase 1: S += scale * q k^T ----
  v16h aq[3];
#pragma unroll
  for (int mt = 0; mt < 3; ++mt)
    aq[mt] = load_frag(qbase, m0 + mt * 16 + r16, hf, 576);
  for (int nt = wave; nt < 9; nt += 4) {
    v16h b = load_frag(kbase, nt * 16 + r16, hf, 576);
    int ncol = nt * 16 + r16;
#pragma unroll
    for (int mt = 0; mt < 3; ++mt) {
      v8f acc = {};
      acc = wmma_f16(aq[mt], b, acc);
#pragma unroll
      for (int e = 0; e < 8; ++e) {
        int mloc = mt * 16 + e + 8 * hf;
        S[mloc * SSTR + ncol] += acc[e] * SCALEF;
      }
    }
  }
  __syncthreads();

  // ---- phase 2: row softmax -> Pf (f16, zero-padded to 160) ----
  for (int r = tid; r < 48; r += 128) {
    float* row = S + r * SSTR;
    float mx = -1e30f;
    for (int j = 0; j < TT; ++j) mx = fmaxf(mx, row[j]);
    float sum = 0.f;
    for (int j = 0; j < TT; ++j) { float ev = __expf(row[j] - mx); row[j] = ev; sum += ev; }
    float inv = 1.f / sum;
    _Float16* prow = Pf + r * 160;
    for (int j = 0; j < TT; ++j) prow[j] = (_Float16)(row[j] * inv);
    for (int j = TT; j < 160; ++j) prow[j] = (_Float16)0.f;
  }
  __syncthreads();

  // ---- phase 3: O = P @ V (48 x 32), K padded to 160 ----
  for (int tile = wave; tile < 6; tile += 4) {
    int mt = tile >> 1, nt = tile & 1;
    v8f acc = {};
#pragma unroll
    for (int kt = 0; kt < 5; ++kt) {
      v16h a = load_frag(Pf + (mt * 16) * 160 + kt * 32, r16, hf, 160);
      v16h b = load_frag(Vt + (nt * 16) * 160 + kt * 32, r16, hf, 160);
      acc = wmma_f16(a, b, acc);
    }
#pragma unroll
    for (int e = 0; e < 8; ++e) {
      int t = m0 + mt * 16 + e + 8 * hf;
      attn_out[((size_t)w * TT + t) * CD + h * HS + nt * 16 + r16] = (_Float16)acc[e];
    }
  }
}

// ---------------- layernorm helpers ----------------
static __device__ __forceinline__ void ln_stats(float v, float& mu, float& var) {
  __shared__ float ps[6], pq[6];
  float s = v, sq = v * v;
#pragma unroll
  for (int o = 16; o > 0; o >>= 1) { s += __shfl_xor(s, o, 32); sq += __shfl_xor(sq, o, 32); }
  int wv = threadIdx.x >> 5;
  if ((threadIdx.x & 31) == 0) { ps[wv] = s; pq[wv] = sq; }
  __syncthreads();
  float ts = 0.f, tq = 0.f;
#pragma unroll
  for (int i = 0; i < 6; ++i) { ts += ps[i]; tq += pq[i]; }
  mu = ts * (1.0f / CD);
  var = tq * (1.0f / CD) - mu * mu;
}

// un-window + un-roll + crop + LN + residual; writes h1 (f32) and h1 (f16)
__global__ __launch_bounds__(192) void unwin_ln_kernel(const float* __restrict__ projout,
                                                       const float* __restrict__ x,
                                                       const float* __restrict__ g,
                                                       const float* __restrict__ bta,
                                                       float* __restrict__ h1f,
                                                       _Float16* __restrict__ h1h,
                                                       const int* __restrict__ rollp) {
  int tok = blockIdx.x;
  int c = threadIdx.x;
  int w0 = tok % W0D, hh = (tok / W0D) % HD, z = tok / (W0D * HD);
  int roll = rollp[0];
  int wp = w0 + 3;
  int zr, hr, wr;
  if (roll) { zr = (z + 1) % ZD; hr = (hh + 6) % HD; wr = (wp + 3) % WPD; }
  else      { zr = z; hr = hh; wr = wp; }
  int t = (zr % WZ) * (WH * WWIN) + (hr % WH) * WWIN + (wr % WWIN);
  int w = (zr / WZ) * (NWH * NWW) + (hr / WH) * NWW + (wr / WWIN);
  float v = projout[((size_t)w * TT + t) * CD + c];
  float mu, var;
  ln_stats(v, mu, var);
  float ln = (v - mu) * rsqrtf(var + 1e-5f) * g[c] + bta[c];
  float hv = x[(size_t)tok * CD + c] + ln;
  h1f[(size_t)tok * CD + c] = hv;
  h1h[(size_t)tok * CD + c] = (_Float16)hv;
}

// out = h1 + LN(ff)
__global__ __launch_bounds__(192) void final_ln_kernel(const float* __restrict__ ff,
                                                       const float* __restrict__ h1f,
                                                       const float* __restrict__ g,
                                                       const float* __restrict__ bta,
                                                       float* __restrict__ out) {
  int tok = blockIdx.x;
  int c = threadIdx.x;
  float v = ff[(size_t)tok * CD + c];
  float mu, var;
  ln_stats(v, mu, var);
  float ln = (v - mu) * rsqrtf(var + 1e-5f) * g[c] + bta[c];
  out[(size_t)tok * CD + c] = h1f[(size_t)tok * CD + c] + ln;
}

// ---------------- host launch ----------------
extern "C" void kernel_launch(void* const* d_in, const int* in_sizes, int n_in,
                              void* d_out, int out_size, void* d_ws, size_t ws_size,
                              hipStream_t stream) {
  const float* x      = (const float*)d_in[0];
  const float* qkv_w  = (const float*)d_in[1];
  const float* qkv_b  = (const float*)d_in[2];
  const float* proj_w = (const float*)d_in[3];
  const float* proj_b = (const float*)d_in[4];
  const float* btab   = (const float*)d_in[5];
  const float* n1w    = (const float*)d_in[6];
  const float* n1b    = (const float*)d_in[7];
  const float* fc1w   = (const float*)d_in[8];
  const float* fc1b   = (const float*)d_in[9];
  const float* fc2w   = (const float*)d_in[10];
  const float* fc2b   = (const float*)d_in[11];
  const float* n2w    = (const float*)d_in[12];
  const float* n2b    = (const float*)d_in[13];
  const int*   roll   = (const int*)d_in[14];

  char* p = (char*)d_ws;
  auto alloc = [&](size_t bytes) -> void* {
    void* r = (void*)p;
    p += (bytes + 255) & ~(size_t)255;
    return r;
  };
  _Float16* qkvw16  = (_Float16*)alloc((size_t)3 * CD * CD * 2);
  _Float16* projw16 = (_Float16*)alloc((size_t)CD * CD * 2);
  _Float16* fc1w16  = (_Float16*)alloc((size_t)4 * CD * CD * 2);
  _Float16* fc2w16  = (_Float16*)alloc((size_t)CD * 4 * CD * 2);
  _Float16* xw      = (_Float16*)alloc((size_t)M_ATT * CD * 2);
  _Float16* qkv16   = (_Float16*)alloc((size_t)M_ATT * 3 * CD * 2);
  _Float16* attnout = (_Float16*)alloc((size_t)M_ATT * CD * 2);
  float*    projout = (float*)alloc((size_t)M_ATT * CD * 4);
  float*    h1f     = (float*)alloc((size_t)TOK * CD * 4);
  _Float16* h1h     = (_Float16*)alloc((size_t)TOK * CD * 2);
  _Float16* hidden  = (_Float16*)alloc((size_t)TOK * 4 * CD * 2);
  float*    ffb     = (float*)alloc((size_t)TOK * CD * 4);

  // weight conversions
  cvt_kernel<<<(3 * CD * CD + 255) / 256, 256, 0, stream>>>(qkv_w, qkvw16, 3 * CD * CD);
  cvt_kernel<<<(CD * CD + 255) / 256, 256, 0, stream>>>(proj_w, projw16, CD * CD);
  cvt_kernel<<<(4 * CD * CD + 255) / 256, 256, 0, stream>>>(fc1w, fc1w16, 4 * CD * CD);
  cvt_kernel<<<(4 * CD * CD + 255) / 256, 256, 0, stream>>>(fc2w, fc2w16, 4 * CD * CD);

  // window gather (8 channels per thread)
  {
    long long n = (long long)NW * TT * (CD / 8);
    gather_kernel<<<(unsigned)((n + 255) / 256), 256, 0, stream>>>(x, xw, roll);
  }

  // QKV GEMM: [138240 x 192] * [576 x 192]^T
  gemm_wmma<0><<<dim3(576 / 64, M_ATT / 128), 256, 0, stream>>>(xw, qkvw16, qkv_b, qkv16,
                                                                M_ATT, 3 * CD, CD);
  // attention
  attn_kernel<<<NW * HEADS * 3, 128, 0, stream>>>(qkv16, btab, attnout, roll);

  // proj GEMM -> f32
  gemm_wmma<1><<<dim3(CD / 64, M_ATT / 128), 256, 0, stream>>>(attnout, projw16, proj_b,
                                                               projout, M_ATT, CD, CD);
  // un-window + LN + residual
  unwin_ln_kernel<<<TOK, CD, 0, stream>>>(projout, x, n1w, n1b, h1f, h1h, roll);

  // FFN
  int mblocks = (TOK + 127) / 128;
  gemm_wmma<2><<<dim3(4 * CD / 64, mblocks), 256, 0, stream>>>(h1h, fc1w16, fc1b, hidden,
                                                               TOK, 4 * CD, CD);
  gemm_wmma<1><<<dim3(CD / 64, mblocks), 256, 0, stream>>>(hidden, fc2w16, fc2b, ffb,
                                                           TOK, CD, 4 * CD);
  // final LN + residual -> out
  final_ln_kernel<<<TOK, CD, 0, stream>>>(ffb, h1f, n2w, n2b, (float*)d_out);
}
